// CROSSATF_27264452395102
// MI455X (gfx1250) — compile-verified, matrix operand order checked by reading, not run on hardware
//
#include <hip/hip_runtime.h>
#include <hip/hip_bf16.h>

// ---------------------------------------------------------------------------
// CDNA5 / gfx1250 implementation. wave32. WMMA bf16 16x16x32 (f32 accum),
// TDM (tensor_load_to_lds) for the GEMM A-panel, LDS radix-2 FFTs.
// LDS tiles are stored fragment-major so each lane reads its WMMA operands
// as contiguous ds_load_b128s.
// ---------------------------------------------------------------------------

typedef __attribute__((ext_vector_type(16))) __bf16 v16bf;
typedef __attribute__((ext_vector_type(8)))  __bf16 v8bf;
typedef __attribute__((ext_vector_type(8)))  float  v8f;
typedef __attribute__((ext_vector_type(4)))  float  v4f;
typedef __attribute__((ext_vector_type(4)))  unsigned int u32x4;
typedef __attribute__((ext_vector_type(8)))  int          i32x8;
typedef __attribute__((ext_vector_type(4)))  int          i32x4;

#define EPSV 1e-5f

#if defined(__has_builtin)
#  if __has_builtin(__builtin_amdgcn_tensor_load_to_lds) && \
      __has_builtin(__builtin_amdgcn_s_wait_tensorcnt)
#    define HAVE_TDM 1
#  endif
#endif
#ifndef HAVE_TDM
#  define HAVE_TDM 0
#endif

union FragAB { v16bf v; v8bf h8[2]; __bf16 h[16]; };
union FragC  { v8f   v; float  f[8];  };

// 16-bit 16x32 A fragment (B 32x16 mirrors it): element e of lane L maps to
// k = (e<8 ? kb+e : 16+kb+e-8), kb = (L>=16)*8. So a fragment is two
// contiguous 8-element runs: [kb, kb+8) and [16+kb, 16+kb+8).
__device__ __forceinline__ void cvt_frag_f32(FragAB& f, const float* p0,
                                             const float* p1) {
  v4f a0 = *(const v4f*)p0;
  v4f a1 = *(const v4f*)(p0 + 4);
  v4f a2 = *(const v4f*)p1;
  v4f a3 = *(const v4f*)(p1 + 4);
#pragma unroll
  for (int e = 0; e < 4; ++e) {
    f.h[e]      = (__bf16)a0[e];
    f.h[e + 4]  = (__bf16)a1[e];
    f.h[e + 8]  = (__bf16)a2[e];
    f.h[e + 12] = (__bf16)a3[e];
  }
}

#if HAVE_TDM
// Issue a TDM load of a 2-D f32 tile (rows x cols, row stride in elements)
// from global memory into LDS. All inputs are wave-uniform. cpol = 0.
__device__ __forceinline__ void tdm_load_tile_f32(
    const float* gptr, void* ldsptr, int cols, int rows, int row_stride)
{
  unsigned long long ga = (unsigned long long)(size_t)gptr;
  unsigned lds_addr = (unsigned)(size_t)ldsptr;     // flat addr[31:0] == LDS byte addr
  const unsigned TD = 1u << 20;                     // oversize tensor dims: no OOB clip
  u32x4 g0;
  g0.x = 1u;                                        // count=1 (valid descriptor)
  g0.y = lds_addr;                                  // D#.lds_addr (bytes)
  g0.z = (unsigned)(ga & 0xFFFFFFFFull);            // global_addr[95:64]
  g0.w = (unsigned)((ga >> 32) & 0x01FFFFFFull) | (2u << 30);   // addr[56:32] | type=2
  i32x8 g1;
  g1[0] = (int)(2u << 16);                          // data_size=2 (4 bytes)
  g1[1] = (int)((TD & 0xFFFFu) << 16);              // tensor_dim0[15:0]
  g1[2] = (int)(((TD >> 16) & 0xFFFFu) | ((TD & 0xFFFFu) << 16));   // td0 hi | td1 lo
  g1[3] = (int)(((TD >> 16) & 0xFFFFu) | ((unsigned)cols << 16));   // td1 hi | tile_dim0
  g1[4] = rows;                                     // tile_dim1 | tile_dim2=0
  g1[5] = row_stride;                               // tensor_dim0_stride[31:0]
  g1[6] = 0;                                        // stride0 hi | stride1 lo
  g1[7] = 0;
  i32x4 gz = {0, 0, 0, 0};
#if defined(__clang_major__) && (__clang_major__ >= 23)
  i32x8 gz8 = {0, 0, 0, 0, 0, 0, 0, 0};
  __builtin_amdgcn_tensor_load_to_lds(g0, g1, gz, gz, gz8, 0);
#else
  __builtin_amdgcn_tensor_load_to_lds(g0, g1, gz, gz, 0);
#endif
}
#endif

// ---------------------------------------------------------------------------
// Generic conv1x1 GEMM: Out[b,m,pos] = epi( sum_k A[m,k] * Bsrc[k, (b,pos)] )
// M = 128 fixed (8 waves x 16 rows). One block = one 16-wide column tile.
// Per K-chunk: wave0 TDM-loads the 128x32 f32 A panel into LDS (overlapped
// with the 256-thread B-tile gather/convert), s_wait_tensorcnt + barrier,
// then 8 WMMAs consume both tiles from LDS via ds_load_b128.
// bmode: 0 = direct (b,Cin,SB) layout, 1 = nearest-upsample x2 gather,
//        2 = zero-padded (OWxOW incl. 1-px border) gather from 64x64 source.
// epi:   0 = none, 1 = BN+ReLU, 2 = +bias
// ---------------------------------------------------------------------------
__global__ __launch_bounds__(256) void gemm128_wmma(
    const float* __restrict__ A, const float* __restrict__ Bsrc,
    float* __restrict__ Out,
    int K, int bmode, int Cin, int SB, int SOut, int OW, int epi,
    const float* __restrict__ g,  const float* __restrict__ bt,
    const float* __restrict__ mu, const float* __restrict__ var,
    const float* __restrict__ bias)
{
  __shared__ float  ldsA[128 * 32];     // f32 A panel (TDM destination), 16KB
  __shared__ __bf16 ldsB[16 * 32];      // bf16 B tile, N-major [n][k]
  const int tid   = threadIdx.x;
  const int lane  = tid & 31;
  const int wave  = tid >> 5;          // m-tile id (0..7)
  const int ntile = blockIdx.x;
  const int n0    = lane & 15;
  const int kb    = (lane >> 4) << 3;

  FragC acc = {};

  const int kchunks = K >> 5;
  for (int kc = 0; kc < kchunks; ++kc) {
    __syncthreads();
#if HAVE_TDM
    if (wave == 0) {                    // one wave drives the DMA
      tdm_load_tile_f32(A + (kc << 5), ldsA, 32, 128, K);
    }
#endif
    // cooperative B-tile gather: 16(N) x 32(K) elements, f32 -> bf16
    for (int e = tid; e < 512; e += 256) {
      int nn = e >> 5;
      int kk = e & 31;
      int ncol = ntile * 16 + nn;
      int b    = ncol / SOut;
      int pos  = ncol - b * SOut;
      int kg   = (kc << 5) + kk;
      float val;
      if (bmode == 0) {
        val = Bsrc[(b * Cin + kg) * SB + pos];
      } else if (bmode == 1) {          // nearest upsample x2
        int i = pos / OW, j = pos - i * OW;
        int spos = (i >> 1) * (OW >> 1) + (j >> 1);
        val = Bsrc[(b * Cin + kg) * SB + spos];
      } else {                          // zero-pad border of OWxOW grid
        int i = pos / OW, j = pos - i * OW;
        if (i == 0 || j == 0 || i == OW - 1 || j == OW - 1) val = 0.0f;
        else val = Bsrc[(b * Cin + kg) * SB + (i - 1) * 64 + (j - 1)];
      }
      ldsB[nn * 32 + kk] = (__bf16)val;
    }
#if HAVE_TDM
    if (wave == 0) __builtin_amdgcn_s_wait_tensorcnt(0);
#endif
    __syncthreads();

    const int m = (wave << 4) + n0;
    FragAB af, bf_;
#if HAVE_TDM
    cvt_frag_f32(af, &ldsA[m * 32 + kb], &ldsA[m * 32 + 16 + kb]);
#else
    const float* arow = A + m * K + (kc << 5);
    __builtin_prefetch(arow + 32, 0, 0);
    cvt_frag_f32(af, arow + kb, arow + 16 + kb);
#endif
    // B fragment: two contiguous 16B runs -> 2x ds_load_b128
    bf_.h8[0] = *(const v8bf*)&ldsB[n0 * 32 + kb];
    bf_.h8[1] = *(const v8bf*)&ldsB[n0 * 32 + 16 + kb];

    acc.v = __builtin_amdgcn_wmma_f32_16x16x32_bf16(
        false, af.v, false, bf_.v, (short)0, acc.v, false, false);
  }

  // Epilogue. D layout: vgpr r, lanes0-15: M=r,N=lane; lanes16-31: M=r+8.
  const int ncol = ntile * 16 + n0;
  const int b    = ncol / SOut;
  const int pos  = ncol - b * SOut;
  const int mhi  = (lane >> 4) << 3;
#pragma unroll
  for (int r = 0; r < 8; ++r) {
    int m = (wave << 4) + mhi + r;
    float v = acc.f[r];
    if (epi == 1) {
      float sc = g[m] * rsqrtf(var[m] + EPSV);
      v = (v - mu[m]) * sc + bt[m];
      v = fmaxf(v, 0.0f);
    } else if (epi == 2) {
      v += bias[m];
    }
    Out[(b * 128 + m) * SOut + pos] = v;
  }
}

// ---------------------------------------------------------------------------
// 3x3 local softmax attention (GROUPS only changes the reshape; per-channel
// math is identical). k/v buffers are in padded 66x66 layout.
// ---------------------------------------------------------------------------
__global__ __launch_bounds__(256) void attn_kernel(
    const float* __restrict__ q, const float* __restrict__ kb,
    const float* __restrict__ vb, const float* __restrict__ rel_h,
    const float* __restrict__ rel_w, float* __restrict__ out)
{
  int idx = blockIdx.x * 256 + threadIdx.x;    // < 4*128*64*64
  int b   = idx >> 19;
  int rem = idx & 524287;
  int c   = rem >> 12;
  int pos = rem & 4095;
  int h   = pos >> 6, w = pos & 63;
  float qv = q[idx];
  int base = (b * 128 + c) * 4356 + h * 66 + w;
  float l[9], vv[9];
  float mx = -3.4e38f;
#pragma unroll
  for (int ki = 0; ki < 3; ++ki)
#pragma unroll
    for (int kj = 0; kj < 3; ++kj) {
      int t = ki * 3 + kj;
      float rel = (c < 64) ? rel_h[c * 3 + ki] : rel_w[(c - 64) * 3 + kj];
      float kk  = kb[base + ki * 66 + kj] + rel;
      l[t]  = qv * kk;
      vv[t] = vb[base + ki * 66 + kj];
      mx = fmaxf(mx, l[t]);
    }
  float s = 0.f, o = 0.f;
#pragma unroll
  for (int t = 0; t < 9; ++t) {
    float e = __expf(l[t] - mx);
    s += e; o += e * vv[t];
  }
  out[idx] = o / s;
}

// ---------------------------------------------------------------------------
// In-place radix-2 LDS FFT along a strided axis (ortho norm per axis).
// Each thread owns one row in its private LDS slice (T*N*8 = 32KB).
// realsrc != nullptr: read real input (imag = 0) from same flat indexing.
// ---------------------------------------------------------------------------
template <int N, int T>
__global__ __launch_bounds__(T) void fft_kernel(
    float2* __restrict__ buf, const float* __restrict__ realsrc,
    int stride, int outer_stride, int inner_count, int inner_stride,
    int nrows, int inverse)
{
  constexpr int LOG2N = (N == 128) ? 7 : 6;
  __shared__ float2 lds[T * N];
  int row = blockIdx.x * T + (int)threadIdx.x;
  if (row >= nrows) return;
  float2* s = lds + threadIdx.x * N;
  int outer = row / inner_count;
  int inner = row - outer * inner_count;
  int base  = outer * outer_stride + inner * inner_stride;

  for (int i = 0; i < N; ++i) {
    int idx = base + i * stride;
    float2 z;
    if (realsrc) { z.x = realsrc[idx]; z.y = 0.f; }
    else         { z = buf[idx]; }
    int rv = (int)(__brev((unsigned)i) >> (32 - LOG2N));
    s[rv] = z;
  }
  const float sgn = inverse ? 1.0f : -1.0f;
  for (int len = 2; len <= N; len <<= 1) {
    float ang = sgn * 6.28318530717958647692f / (float)len;
    int half = len >> 1;
    for (int i = 0; i < N; i += len)
      for (int j = 0; j < half; ++j) {
        float sn, cs;
        __sincosf(ang * (float)j, &sn, &cs);
        float2 u = s[i + j];
        float2 t = s[i + j + half];
        float2 wt = make_float2(cs * t.x - sn * t.y, cs * t.y + sn * t.x);
        s[i + j]        = make_float2(u.x + wt.x, u.y + wt.y);
        s[i + j + half] = make_float2(u.x - wt.x, u.y - wt.y);
      }
  }
  const float nrm = 1.0f / sqrtf((float)N);
  for (int i = 0; i < N; ++i)
    buf[base + i * stride] = make_float2(s[i].x * nrm, s[i].y * nrm);
}

// Length-4 ortho DFT along the batch axis (stride 524288 complex elements).
__global__ __launch_bounds__(256) void dft4_kernel(float2* __restrict__ ft)
{
  int p = blockIdx.x * 256 + threadIdx.x;      // < 128*64*64
  const int S = 524288;
  float2 z0 = ft[p], z1 = ft[p + S], z2 = ft[p + 2 * S], z3 = ft[p + 3 * S];
  float2 e0 = make_float2(z0.x + z2.x, z0.y + z2.y);
  float2 o0 = make_float2(z0.x - z2.x, z0.y - z2.y);
  float2 e1 = make_float2(z1.x + z3.x, z1.y + z3.y);
  float2 o1 = make_float2(z1.x - z3.x, z1.y - z3.y);
  const float n = 0.5f;
  ft[p]         = make_float2((e0.x + e1.x) * n, (e0.y + e1.y) * n);
  ft[p + S]     = make_float2((o0.x + o1.y) * n, (o0.y - o1.x) * n); // o0 - i*o1
  ft[p + 2 * S] = make_float2((e0.x - e1.x) * n, (e0.y - e1.y) * n);
  ft[p + 3 * S] = make_float2((o0.x - o1.y) * n, (o0.y + o1.x) * n); // o0 + i*o1
}

// ---------------------------------------------------------------------------
// Per-frequency complex GEMM: out[b,o] = sum_i ft[b,i] * w1[i,o]  (in-place),
// fused with the fftshift/mask/ifftshift (= zero freqs f_h,f_w in {-3..2};
// the channel-axis shift cancels). One block per (x,y); 8 waves = 8 o-tiles;
// A = ft rows (b padded 4->16), staged in LDS [b][i] so a lane's fragment is
// contiguous. 4 real bf16 WMMAs per K-chunk of 32 (Re/Im accumulators;
// -Ai*Bi via pre-negated B fragment since NEG[1:0] is reserved for bf16 WMMA).
// ---------------------------------------------------------------------------
__global__ __launch_bounds__(256) void spectral_wmma(
    float2* __restrict__ ft, const float* __restrict__ w1r,
    const float* __restrict__ w1i)
{
  __shared__ float aRe[512];   // [b*128 + i]
  __shared__ float aIm[512];
  const int xy    = blockIdx.x;          // 0..4095
  const int x     = xy >> 6, y = xy & 63;
  const int tid   = threadIdx.x;
  const int lane  = tid & 31;
  const int otile = tid >> 5;
  const int n0    = lane & 15;
  const int kb    = (lane >> 4) << 3;

  bool cut = (((x + 3) & 63) < 6) && (((y + 3) & 63) < 6);
  if (cut) {                              // block-uniform: safe early exit
    if (lane < 16) {
      int o = (otile << 4) + lane;
      for (int b = 0; b < 4; ++b)
        ft[((b * 128 + o) << 12) + xy] = make_float2(0.f, 0.f);
    }
    return;
  }

  for (int t = tid; t < 512; t += 256) {  // stage ft[:, :, x, y] in LDS
    int b = t >> 7, i = t & 127;
    float2 z = ft[((b * 128 + i) << 12) + xy];
    aRe[t] = z.x; aIm[t] = z.y;
  }
  __syncthreads();

  FragC cRe = {}, cIm = {};
  const int mrow = n0;                    // b-row (valid < 4)
  const int oo   = (otile << 4) + n0;     // B column = output channel
  for (int kc = 0; kc < 4; ++kc) {
    FragAB ar, ai, br, bi, bin;
    if (mrow < 4) {                       // contiguous i-runs -> ds_load_b128
      const float* pr = &aRe[mrow * 128 + (kc << 5) + kb];
      const float* pi = &aIm[mrow * 128 + (kc << 5) + kb];
      cvt_frag_f32(ar, pr, pr + 16);
      cvt_frag_f32(ai, pi, pi + 16);
    } else {
#pragma unroll
      for (int e = 0; e < 16; ++e) { ar.h[e] = (__bf16)0.f; ai.h[e] = (__bf16)0.f; }
    }
#pragma unroll
    for (int e = 0; e < 16; ++e) {
      int k = (kc << 5) + ((e < 8) ? (kb + e) : (16 + kb + (e - 8)));
      int widx = (k * 128 + oo) * 4096 + xy;
      float wr = w1r[widx], wi = w1i[widx];
      br.h[e]  = (__bf16)wr;
      bi.h[e]  = (__bf16)wi;
      bin.h[e] = (__bf16)(-wi);
    }
    cRe.v = __builtin_amdgcn_wmma_f32_16x16x32_bf16(false, ar.v, false, br.v,  (short)0, cRe.v, false, false);
    cRe.v = __builtin_amdgcn_wmma_f32_16x16x32_bf16(false, ai.v, false, bin.v, (short)0, cRe.v, false, false);
    cIm.v = __builtin_amdgcn_wmma_f32_16x16x32_bf16(false, ar.v, false, bi.v,  (short)0, cIm.v, false, false);
    cIm.v = __builtin_amdgcn_wmma_f32_16x16x32_bf16(false, ai.v, false, br.v,  (short)0, cIm.v, false, false);
  }

  // D rows 0..3 (= b) live in vgprs 0..3 of lanes 0..15 (N = o within tile).
  if (lane < 16) {
    int o = (otile << 4) + lane;
#pragma unroll
    for (int r = 0; r < 4; ++r)
      ft[((r * 128 + o) << 12) + xy] = make_float2(cRe.f[r], cIm.f[r]);
  }
}

// y_out = relu(bn2(rate1 * Re(ifft) + rate2 * out2))
__global__ __launch_bounds__(256) void final_kernel(
    const float2* __restrict__ ft, const float* __restrict__ out2,
    const float* __restrict__ rate1, const float* __restrict__ rate2,
    const float* __restrict__ g,  const float* __restrict__ bt,
    const float* __restrict__ mu, const float* __restrict__ var,
    float* __restrict__ out)
{
  int idx = blockIdx.x * 256 + threadIdx.x;    // < 2097152
  int c = (idx >> 12) & 127;
  float v = rate1[0] * ft[idx].x + rate2[0] * out2[idx];
  float sc = g[c] * rsqrtf(var[c] + EPSV);
  v = (v - mu[c]) * sc + bt[c];
  out[idx] = fmaxf(v, 0.0f);
}

// ---------------------------------------------------------------------------
extern "C" void kernel_launch(void* const* d_in, const int* in_sizes, int n_in,
                              void* d_out, int out_size, void* d_ws, size_t ws_size,
                              hipStream_t stream)
{
  (void)in_sizes; (void)n_in; (void)out_size; (void)ws_size;
  const float* x      = (const float*)d_in[0];
  const float* y      = (const float*)d_in[1];
  const float* conv1w = (const float*)d_in[2];
  const float* bn1g   = (const float*)d_in[3];
  const float* bn1b   = (const float*)d_in[4];
  const float* bn1m   = (const float*)d_in[5];
  const float* bn1v   = (const float*)d_in[6];
  const float* qw     = (const float*)d_in[7];
  const float* kw     = (const float*)d_in[8];
  const float* vw     = (const float*)d_in[9];
  const float* relh   = (const float*)d_in[10];
  const float* relw   = (const float*)d_in[11];
  const float* rate1  = (const float*)d_in[12];
  const float* rate2  = (const float*)d_in[13];
  const float* w1r    = (const float*)d_in[14];
  const float* w1i    = (const float*)d_in[15];
  const float* w0w    = (const float*)d_in[16];
  const float* w0b    = (const float*)d_in[17];
  const float* bn2g   = (const float*)d_in[18];
  const float* bn2b   = (const float*)d_in[19];
  const float* bn2m   = (const float*)d_in[20];
  const float* bn2v   = (const float*)d_in[21];

  // Workspace layout (floats); lifetimes allow reuse -> 7081984 floats (~28MB)
  float*  ws   = (float*)d_ws;
  float*  y1   = ws;                       // 524288   (4,128,32,32)
  float*  q    = ws + 524288;              // 2097152  (4,128,64,64)
  float*  kbuf = ws + 2621440;             // 2230272  (4,128,66,66)
  float*  vbuf = ws + 4851712;             // 2230272  (4,128,66,66)
  float2* ft   = (float2*)(ws + 2621440);  // 2097152 cplx, reuses k/v region
  float*  out2 = ws + 524288;              // 2097152, reuses q region

  float* yout = (float*)d_out;             // first output
  float* xatt = (float*)d_out + 2097152;   // second output

  dim3 blk(256);

  // 1) y1 = relu(bn1(conv1x1(y))) : M=128 K=256 N=4096
  gemm128_wmma<<<dim3(256), blk, 0, stream>>>(conv1w, y, y1,
      256, 0, 256, 1024, 1024, 32, 1, bn1g, bn1b, bn1m, bn1v, nullptr);
  // 2) q = conv1x1(upsample(y1)) : N=16384
  gemm128_wmma<<<dim3(1024), blk, 0, stream>>>(qw, y1, q,
      128, 1, 128, 1024, 4096, 64, 0, nullptr, nullptr, nullptr, nullptr, nullptr);
  // 3,4) k,v on zero-padded x : N=4*66*66=17424
  gemm128_wmma<<<dim3(1089), blk, 0, stream>>>(kw, x, kbuf,
      128, 2, 128, 4096, 4356, 66, 0, nullptr, nullptr, nullptr, nullptr, nullptr);
  gemm128_wmma<<<dim3(1089), blk, 0, stream>>>(vw, x, vbuf,
      128, 2, 128, 4096, 4356, 66, 0, nullptr, nullptr, nullptr, nullptr, nullptr);
  // 5) attention -> x_att (second half of d_out)
  attn_kernel<<<dim3(8192), blk, 0, stream>>>(q, kbuf, vbuf, relh, relw, xatt);
  // 6) forward 4-D ortho FFT of x_att into ft: w, h, c, b axes
  fft_kernel<64, 64><<<dim3(512), dim3(64), 0, stream>>>(ft, xatt, 1, 64, 1, 0, 32768, 0);
  fft_kernel<64, 64><<<dim3(512), dim3(64), 0, stream>>>(ft, nullptr, 64, 4096, 64, 1, 32768, 0);
  fft_kernel<128, 32><<<dim3(512), dim3(32), 0, stream>>>(ft, nullptr, 4096, 524288, 4096, 1, 16384, 0);
  dft4_kernel<<<dim3(2048), blk, 0, stream>>>(ft);
  // 7) per-frequency complex GEMM with w1 + ring mask (in-place)
  spectral_wmma<<<dim3(4096), blk, 0, stream>>>(ft, w1r, w1i);
  // 8) inverse ortho FFT over (h,w) only
  fft_kernel<64, 64><<<dim3(512), dim3(64), 0, stream>>>(ft, nullptr, 1, 64, 1, 0, 32768, 1);
  fft_kernel<64, 64><<<dim3(512), dim3(64), 0, stream>>>(ft, nullptr, 64, 4096, 64, 1, 32768, 1);
  // 9) out2 = conv1x1(x_att) + w0_b
  gemm128_wmma<<<dim3(1024), blk, 0, stream>>>(w0w, xatt, out2,
      128, 0, 128, 4096, 4096, 64, 2, nullptr, nullptr, nullptr, nullptr, w0b);
  // 10) y_out = relu(bn2(rate1*Re + rate2*out2))
  final_kernel<<<dim3(8192), blk, 0, stream>>>(ft, out2, rate1, rate2,
      bn2g, bn2b, bn2m, bn2v, yout);
}
